// OrchidOperator_24644522345244
// MI455X (gfx1250) — matrix operand order repaired and checked by reading, not compiled, not run on hardware
//
#include <hip/hip_runtime.h>
#include <hip/hip_bf16.h>
#include <math.h>

// ---------------------------------------------------------------------------
// Problem constants (from setup_inputs)
// ---------------------------------------------------------------------------
#define BB 4
#define SS 8192
#define DD 768
#define WW (3 * DD)
#define FF (SS / 2 + 1)   // 4097

typedef __attribute__((ext_vector_type(16))) __bf16 v16bf;
typedef __attribute__((ext_vector_type(4)))  __bf16 v4bf;
typedef __attribute__((ext_vector_type(8)))  float  v8f;
typedef __attribute__((ext_vector_type(4)))  unsigned int u32x4;
typedef __attribute__((ext_vector_type(4)))  int    i32x4;
typedef __attribute__((ext_vector_type(8)))  int    i32x8;

#if defined(__has_builtin)
#if __has_builtin(__builtin_amdgcn_tensor_load_to_lds)
#define HAS_TDM 1
#endif
#endif

// ===========================================================================
// bf16 GEMM with TDM-staged LDS tiles:
//   C[M,N] = A[M,K](bf16,rm) * Wt[N,K](bf16,rm)^T + bias[N]
//   block = 256 thr / 8 waves; C tile 128x128; wave tile 32x64 (2x4 frags)
//   K step 32; tiles DMA'd by the Tensor Data Mover, double buffered,
//   with 4-dword LDS pad every 16 dwords (row stride 40 bf16 = 80B).
// ===========================================================================
#define GBM 128
#define GBN 128
#define GBK 32
#define LROW 40           // bf16 per padded LDS row (80 bytes, 16B aligned)

#ifdef HAS_TDM
// Issue one TDM 2D-tile load: rows x GBK bf16 tile at (row0, k0) of a
// row-major (nrows_total, ldk) bf16 matrix -> LDS at lds_off (padded layout).
__device__ __forceinline__ void tdm_tile_load(const __bf16* gbase, int ldk,
                                              int row0, int k0, unsigned lds_off)
{
    unsigned long long ga =
        (unsigned long long)(uintptr_t)(gbase + (size_t)row0 * ldk + k0);
    u32x4 g0;
    g0[0] = 1u;                                        // count=1, user mode
    g0[1] = lds_off;                                   // LDS byte address
    g0[2] = (unsigned)(ga & 0xffffffffu);              // global_addr[31:0]
    g0[3] = (unsigned)((ga >> 32) & 0x01ffffffu)       // global_addr[56:32]
          | (2u << 30);                                // type = 2 (image)
    i32x8 g1;
    g1[0] = (1 << 16)                                  // data_size = 2 bytes
          | (1 << 20)                                  // pad_enable
          | (3 << 22)                                  // pad_interval: 16 dwords
          | (3 << 25);                                 // pad_amount: 4 dwords
    g1[1] = (ldk & 0xffff) << 16;                      // tensor_dim0 lo16
    g1[2] = ((unsigned)ldk >> 16)                      // tensor_dim0 hi16
          | (0u << 16);                                // tensor_dim1 lo16
    g1[3] = (0x0010)                                   // tensor_dim1 hi16 (=1<<20 rows)
          | (GBK << 16);                               // tile_dim0 = 32 elems
    g1[4] = GBM | (0 << 16);                           // tile_dim1 = 128, tile_dim2 = 0
    g1[5] = ldk;                                       // tensor_dim0_stride lo32
    g1[6] = 0;                                         // stride hi16 | dim1_stride lo
    g1[7] = 0;
    i32x4 gz = {0, 0, 0, 0};                           // groups 2/3: dims unused
#if __clang_major__ >= 23
    i32x8 gz8 = {0, 0, 0, 0, 0, 0, 0, 0};
    __builtin_amdgcn_tensor_load_to_lds(g0, g1, gz, gz, gz8, 0);
#else
    __builtin_amdgcn_tensor_load_to_lds(g0, g1, gz, gz, 0);
#endif
}
#endif

// manual staging fallback / helper: copy tile into padded LDS layout, 16B chunks
__device__ __forceinline__ void manual_tile_load(const __bf16* gbase, int ldk,
                                                 int row0, int k0, __bf16* buf, int tid)
{
    // GBM*GBK/8 = 512 16-byte chunks, 256 threads -> 2 chunks each
    #pragma unroll
    for (int c = tid; c < GBM * GBK / 8; c += 256) {
        int r = c >> 2, c8 = (c & 3) * 8;
        *(uint4*)&buf[r * LROW + c8] =
            *(const uint4*)&gbase[(size_t)(row0 + r) * ldk + k0 + c8];
    }
}

__global__ __launch_bounds__(256)
void gemm_bf16_wmma(const __bf16* __restrict__ A, const __bf16* __restrict__ Wt,
                    const float* __restrict__ bias, float* __restrict__ C,
                    int M, int N, int K)
{
    __shared__ __bf16 Atile[2][GBM * LROW];   // 2 x 10 KB
    __shared__ __bf16 Btile[2][GBN * LROW];   // 2 x 10 KB

    const int tid  = threadIdx.x;
    const int lane = tid & 31;
    const int wave = tid >> 5;
    const int wm   = wave & 3;            // 4 waves along M (32 rows each)
    const int wn   = wave >> 2;           // 2 waves along N (64 cols each)
    const int lrow = lane & 15;
    const int lhi  = lane >> 4;

    const int m0 = blockIdx.y * GBM;
    const int n0 = blockIdx.x * GBN;
    const int ntiles = K / GBK;

    v8f acc[2][4] = {};

#ifdef HAS_TDM
    if (wave == 0) {
        tdm_tile_load(A,  K, m0, 0, (unsigned)(uintptr_t)&Atile[0][0]);
        tdm_tile_load(Wt, K, n0, 0, (unsigned)(uintptr_t)&Btile[0][0]);
    }
#endif

    for (int t = 0; t < ntiles; ++t) {
        const int cur = t & 1;
#ifdef HAS_TDM
        if (wave == 0) {
            if (t + 1 < ntiles) {
                // prefetch next tile into the other buffer, then wait for the
                // two DMAs of the CURRENT tile (in-order completion) while the
                // next two stay in flight.
                tdm_tile_load(A,  K, m0, (t + 1) * GBK,
                              (unsigned)(uintptr_t)&Atile[cur ^ 1][0]);
                tdm_tile_load(Wt, K, n0, (t + 1) * GBK,
                              (unsigned)(uintptr_t)&Btile[cur ^ 1][0]);
                __builtin_amdgcn_s_wait_tensorcnt((short)2);
            } else {
                __builtin_amdgcn_s_wait_tensorcnt((short)0);
            }
        }
        __syncthreads();
#else
        manual_tile_load(A,  K, m0, t * GBK, &Atile[cur][0], tid);
        manual_tile_load(Wt, K, n0, t * GBK, &Btile[cur][0], tid);
        __syncthreads();
#endif

        // --- fragment gathers (ISA 7.12.2 16-bit layouts), two b128s each ---
        v16bf afr[2], bfr[4];
        #pragma unroll
        for (int fm = 0; fm < 2; ++fm) {
            const __bf16* rp = &Atile[cur][(wm * 32 + fm * 16 + lrow) * LROW];
            #pragma unroll
            for (int j = 0; j < 16; ++j)
                afr[fm][j] = rp[j + ((j >= 8) ? 8 : 0) + 8 * lhi];
        }
        #pragma unroll
        for (int fn = 0; fn < 4; ++fn) {
            const __bf16* rp = &Btile[cur][(wn * 64 + fn * 16 + lrow) * LROW];
            #pragma unroll
            for (int j = 0; j < 16; ++j)
                bfr[fn][j] = rp[j + ((j >= 8) ? 8 : 0) + 8 * lhi];
        }
        #pragma unroll
        for (int fm = 0; fm < 2; ++fm)
            #pragma unroll
            for (int fn = 0; fn < 4; ++fn)
                acc[fm][fn] = __builtin_amdgcn_wmma_f32_16x16x32_bf16(
                    false, afr[fm], false, bfr[fn],
                    (short)0, acc[fm][fn], false, false);
        __syncthreads();   // everyone done reading buf[cur] before its reuse
    }

    // --- store: VGPR r holds M = r + 8*lhi ; N = lrow ---
    #pragma unroll
    for (int fm = 0; fm < 2; ++fm) {
        #pragma unroll
        for (int fn = 0; fn < 4; ++fn) {
            int n = n0 + wn * 64 + fn * 16 + lrow;
            float bv = bias[n];
            #pragma unroll
            for (int r = 0; r < 8; ++r) {
                int m = m0 + wm * 32 + fm * 16 + r + 8 * lhi;
                C[(size_t)m * N + n] = acc[fm][fn][r] + bv;
            }
        }
    }
}

// ===========================================================================
// fp32 -> bf16 bulk convert (vectorized: float4 in, 8B out)
// ===========================================================================
__global__ __launch_bounds__(256)
void cvt_f32_bf16(const float* __restrict__ in, __bf16* __restrict__ out, long n4)
{
    long i = (long)blockIdx.x * 256 + threadIdx.x;
    if (i >= n4) return;
    float4 f = ((const float4*)in)[i];
    v4bf p;
    p[0] = (__bf16)f.x; p[1] = (__bf16)f.y; p[2] = (__bf16)f.z; p[3] = (__bf16)f.w;
    ((v4bf*)out)[i] = p;
}

// ===========================================================================
// Fused depthwise convs over S:  from x3 (B,S,W) produce (B,D,S):
//   yt = shortconv(vv)*shortconv(s1);  s2t = shortconv(s2);
//   hc = dwconv(raw v, w_cond1)+b_cond1
// ===========================================================================
__global__ __launch_bounds__(256)
void fused_dwconv(const float* __restrict__ x3,
                  const float* __restrict__ w_short, const float* __restrict__ b_short,
                  const float* __restrict__ w_cond1, const float* __restrict__ b_cond1,
                  float* __restrict__ yt, float* __restrict__ s2t, float* __restrict__ hc)
{
    const int b = blockIdx.z;
    const int s = blockIdx.x * 32 + threadIdx.x;
    const int d = blockIdx.y * 8  + threadIdx.y;

    float v[3][3];
    #pragma unroll
    for (int t = 0; t < 3; ++t) {
        int ss = s + t - 1;
        bool ok = (ss >= 0) && (ss < SS);
        size_t base = ((size_t)b * SS + (ok ? ss : 0)) * WW;
        #pragma unroll
        for (int g = 0; g < 3; ++g)
            v[t][g] = ok ? x3[base + g * DD + d] : 0.0f;
    }

    float s1 = b_short[d], s2 = b_short[DD + d], vv = b_short[2 * DD + d];
    float hcv = b_cond1[d];
    #pragma unroll
    for (int t = 0; t < 3; ++t) {
        s1  += v[t][0] * w_short[(size_t)d * 3 + t];
        s2  += v[t][1] * w_short[(size_t)(DD + d) * 3 + t];
        vv  += v[t][2] * w_short[(size_t)(2 * DD + d) * 3 + t];
        hcv += v[t][2] * w_cond1[(size_t)d * 3 + t];
    }

    size_t idx = ((size_t)b * DD + d) * SS + s;
    yt[idx]  = s1 * vv;
    s2t[idx] = s2;
    hc[idx]  = hcv;
}

// ===========================================================================
// Radix-2 DIT FFT (8192 pts) entirely in LDS (64KB of the 320KB WGP LDS).
// ===========================================================================
__device__ __forceinline__ void fft8192_lds(float* re, float* im, int tid, float sign)
{
    for (int i = tid; i < SS; i += 256) {
        int j = (int)(__brev((unsigned)i) >> 19);
        if (i < j) {
            float tr = re[i]; re[i] = re[j]; re[j] = tr;
            float ti = im[i]; im[i] = im[j]; im[j] = ti;
        }
    }
    __syncthreads();

    for (int st = 1; st <= 13; ++st) {
        const int half = 1 << (st - 1);
        const float ang0 = sign * 6.283185307179586f / (float)(1 << st);
        for (int t = tid; t < (SS >> 1); t += 256) {
            int j = t & (half - 1);
            int p = ((t >> (st - 1)) << st) + j;
            float sn, cs;
            __sincosf(ang0 * (float)j, &sn, &cs);
            float ur = re[p],        ui = im[p];
            float vr = re[p + half], vi = im[p + half];
            float tr = vr * cs - vi * sn;
            float ti = vr * sn + vi * cs;
            re[p]        = ur + tr;  im[p]        = ui + ti;
            re[p + half] = ur - tr;  im[p + half] = ui - ti;
        }
        __syncthreads();
    }
}

__global__ __launch_bounds__(256)
void fft_mag_kernel(const float* __restrict__ hc, float* __restrict__ hmag)
{
    __shared__ float re[SS];
    __shared__ float im[SS];
    const int row = blockIdx.x;
    const int tid = threadIdx.x;
    for (int i = tid; i < SS; i += 256) { re[i] = hc[(size_t)row * SS + i]; im[i] = 0.0f; }
    __syncthreads();
    fft8192_lds(re, im, tid, -1.0f);
    for (int k = tid; k < FF; k += 256)
        hmag[(size_t)row * FF + k] = sqrtf(re[k] * re[k] + im[k] * im[k]);
}

__global__ __launch_bounds__(256)
void freq_dwconv(const float* __restrict__ hmag,
                 const float* __restrict__ w_cond2, const float* __restrict__ b_cond2,
                 float* __restrict__ hadapt)
{
    size_t i = (size_t)blockIdx.x * 256 + threadIdx.x;
    if (i >= (size_t)BB * DD * FF) return;
    int k = (int)(i % FF);
    size_t row = i / FF;
    int d = (int)(row % DD);
    const float* hr = hmag + row * FF;
    float acc = b_cond2[d];
    if (k - 1 >= 0) acc += hr[k - 1] * w_cond2[d * 3 + 0];
    acc += hr[k] * w_cond2[d * 3 + 1];
    if (k + 1 < FF) acc += hr[k + 1] * w_cond2[d * 3 + 2];
    hadapt[i] = acc;
}

__global__ __launch_bounds__(256)
void fft_h0_kernel(const float* __restrict__ h0,
                   float* __restrict__ h0re, float* __restrict__ h0im)
{
    __shared__ float re[SS];
    __shared__ float im[SS];
    const int d = blockIdx.x;
    const int tid = threadIdx.x;
    for (int i = tid; i < SS; i += 256) { re[i] = h0[(size_t)i * DD + d]; im[i] = 0.0f; }
    __syncthreads();
    fft8192_lds(re, im, tid, -1.0f);
    for (int k = tid; k < FF; k += 256) {
        h0re[(size_t)d * FF + k] = re[k];
        h0im[(size_t)d * FF + k] = im[k];
    }
}

__global__ __launch_bounds__(256)
void spectral_kernel(float* __restrict__ yt, const float* __restrict__ s2t,
                     const float* __restrict__ hadapt,
                     const float* __restrict__ h0re, const float* __restrict__ h0im)
{
    __shared__ float re[SS];
    __shared__ float im[SS];
    const int row = blockIdx.x;
    const int d   = row % DD;
    const int tid = threadIdx.x;

    for (int i = tid; i < SS; i += 256) { re[i] = yt[(size_t)row * SS + i]; im[i] = 0.0f; }
    __syncthreads();
    fft8192_lds(re, im, tid, -1.0f);

    for (int k = tid; k < FF; k += 256) {
        float xr = re[k], xi = im[k];
        float hr = h0re[(size_t)d * FF + k] + hadapt[(size_t)row * FF + k];
        float hi = h0im[(size_t)d * FF + k];
        float pr = xr * hr - xi * hi;
        float pi = xr * hi + xi * hr;
        re[k] = pr; im[k] = pi;
        if (k > 0 && k < SS / 2) { re[SS - k] = pr; im[SS - k] = -pi; }
    }
    __syncthreads();

    fft8192_lds(re, im, tid, +1.0f);

    const float inv = 1.0f / (float)SS;
    for (int i = tid; i < SS; i += 256)
        yt[(size_t)row * SS + i] = re[i] * inv * s2t[(size_t)row * SS + i];
}

// ===========================================================================
// transpose (B,D,S) fp32 -> (B,S,D) bf16 (A operand of the output GEMM)
// ===========================================================================
__global__ __launch_bounds__(256)
void transpose_ds_bf16(const float* __restrict__ in, __bf16* __restrict__ out)
{
    __shared__ float tile[32][33];
    const int b  = blockIdx.z;
    const int s0 = blockIdx.x * 32;
    const int d0 = blockIdx.y * 32;
    for (int i = threadIdx.y; i < 32; i += 8)
        tile[i][threadIdx.x] = in[((size_t)b * DD + d0 + i) * SS + s0 + threadIdx.x];
    __syncthreads();
    for (int i = threadIdx.y; i < 32; i += 8)
        out[((size_t)b * SS + s0 + i) * DD + d0 + threadIdx.x] =
            (__bf16)tile[threadIdx.x][i];
}

// ===========================================================================
// Host-side launch
// ===========================================================================
extern "C" void kernel_launch(void* const* d_in, const int* in_sizes, int n_in,
                              void* d_out, int out_size, void* d_ws, size_t ws_size,
                              hipStream_t stream)
{
    const float* x       = (const float*)d_in[0];
    const float* w_in    = (const float*)d_in[1];
    const float* b_in    = (const float*)d_in[2];
    const float* w_short = (const float*)d_in[3];
    const float* b_short = (const float*)d_in[4];
    const float* w_cond1 = (const float*)d_in[5];
    const float* b_cond1 = (const float*)d_in[6];
    const float* w_cond2 = (const float*)d_in[7];
    const float* b_cond2 = (const float*)d_in[8];
    const float* h0      = (const float*)d_in[9];
    const float* w_out   = (const float*)d_in[10];
    const float* b_out   = (const float*)d_in[11];
    float* out = (float*)d_out;

    // workspace layout
    float* ws = (float*)d_ws;
    size_t off = 0;
    float* x3     = ws + off; off += (size_t)BB * SS * WW;   // dead after conv pass
    float* yt     = ws + off; off += (size_t)BB * DD * SS;
    float* s2t    = ws + off; off += (size_t)BB * DD * SS;
    float* hc     = ws + off; off += (size_t)BB * DD * SS;
    float* hmag   = ws + off; off += (size_t)BB * DD * FF;
    float* hadapt = ws + off; off += (size_t)BB * DD * FF;
    float* h0re   = ws + off; off += (size_t)DD * FF;
    float* h0im   = ws + off; off += (size_t)DD * FF;
    __bf16* xb    = (__bf16*)(ws + off);
    off += ((size_t)BB * SS * DD + 1) / 2;
    __bf16* w_inb = (__bf16*)(ws + off);
    off += ((size_t)WW * DD + 1) / 2;
    __bf16* w_outb = (__bf16*)(ws + off);
    off += ((size_t)DD * DD + 1) / 2;
    __bf16* y2b   = (__bf16*)x3;                             // reuse dead x3 region

    const int M = BB * SS;

    // 0) one-time bf16 conversions (activations + both weight matrices)
    {
        long n4 = (long)BB * SS * DD / 4;
        cvt_f32_bf16<<<(unsigned)((n4 + 255) / 256), 256, 0, stream>>>(x, xb, n4);
        n4 = (long)WW * DD / 4;
        cvt_f32_bf16<<<(unsigned)((n4 + 255) / 256), 256, 0, stream>>>(w_in, w_inb, n4);
        n4 = (long)DD * DD / 4;
        cvt_f32_bf16<<<(unsigned)((n4 + 255) / 256), 256, 0, stream>>>(w_out, w_outb, n4);
    }

    // 1) x3 = x @ w_in^T + b_in   (TDM-staged WMMA GEMM)
    gemm_bf16_wmma<<<dim3(WW / GBN, M / GBM), 256, 0, stream>>>(
        xb, w_inb, b_in, x3, M, WW, DD);

    // 2) fused short/cond1 depthwise convs -> yt, s2t, hc  (all (B,D,S))
    fused_dwconv<<<dim3(SS / 32, DD / 8, BB), dim3(32, 8), 0, stream>>>(
        x3, w_short, b_short, w_cond1, b_cond1, yt, s2t, hc);

    // 3) hmag = |rfft(hc)|
    fft_mag_kernel<<<BB * DD, 256, 0, stream>>>(hc, hmag);

    // 4) hadapt = dwconv_freq(hmag)
    {
        size_t tot = (size_t)BB * DD * FF;
        freq_dwconv<<<(unsigned)((tot + 255) / 256), 256, 0, stream>>>(
            hmag, w_cond2, b_cond2, hadapt);
    }

    // 5) h0_f = rfft(h0_t)
    fft_h0_kernel<<<DD, 256, 0, stream>>>(h0, h0re, h0im);

    // 6) yt = irfft(rfft(yt) * (h0_f + hadapt)) * s2t   (in place)
    spectral_kernel<<<BB * DD, 256, 0, stream>>>(yt, s2t, hadapt, h0re, h0im);

    // 7) y2b (B,S,D bf16) = transpose(yt)
    transpose_ds_bf16<<<dim3(SS / 32, DD / 32, BB), dim3(32, 8), 0, stream>>>(yt, y2b);

    // 8) out = y2 @ w_out^T + b_out
    gemm_bf16_wmma<<<dim3(DD / GBN, M / GBM), 256, 0, stream>>>(
        y2b, w_outb, b_out, out, M, DD, DD);
}
